// MoEGate_47081431499148
// MI455X (gfx1250) — matrix operand order repaired and checked by reading, not compiled, not run on hardware
//
#include <hip/hip_runtime.h>
#include <hip/hip_bf16.h>

// ---- problem constants (from reference) ----
#define NE      64          // experts
#define TOPK    8
#define BB      4           // batch
#define SS      4096        // seq len
#define HH      2048        // hidden
#define NTOK    (BB*SS)     // 16384 tokens
#define ALPHA   0.1f

// ---- tiling ----
#define MT      64          // tokens per block (4 waves x 16 rows)
#define NTHREADS 128
#define KCH     512         // K-chunk of W resident in LDS per stage
#define WPITCH  520         // bf16 per W row in LDS (512 + 8 pad -> 16B pad, conflict-free b128 reads)
#define LOGP    65          // floats per logits row in LDS (64 + 1 pad)

#define W_LDS_BYTES   (NE * WPITCH * 2)          // 66560
#define LOG_LDS_BYTES (MT * LOGP * 4)            // 16640
#define CNT_LDS_BYTES (NE * 4)                   // 256
#define SUM_LDS_BYTES (NE * 4)                   // 256
#define SMEM_BYTES    (W_LDS_BYTES + LOG_LDS_BYTES + CNT_LDS_BYTES + SUM_LDS_BYTES) // 83712 -> 3 WGs/WGP

typedef __attribute__((ext_vector_type(16))) __bf16 v16bf;
typedef __attribute__((ext_vector_type(8)))  float  v8f;

// ------------------------------------------------------------------
// zero the aux-loss accumulators in workspace: counts[B*E], sums[B*E]
// ------------------------------------------------------------------
__global__ __launch_bounds__(256)
void moe_gate_zero(float* ws) {
    int t = threadIdx.x;
    ws[t]       = 0.0f;   // counts [4*64]
    ws[t + 256] = 0.0f;   // score sums [4*64]
}

// ------------------------------------------------------------------
// fused: logits GEMM (WMMA bf16, K-chunked W in LDS) + softmax +
//        top-8 + aux accumulation
// ------------------------------------------------------------------
__global__ __launch_bounds__(NTHREADS)
void moe_gate_main(const float* __restrict__ x,       // [NTOK, HH]
                   const float* __restrict__ w,       // [NE, HH]
                   float* __restrict__ out_idx,       // [NTOK, TOPK] (indices as float)
                   float* __restrict__ out_w,         // [NTOK, TOPK]
                   float* __restrict__ g_counts,      // [BB, NE]
                   float* __restrict__ g_sums)        // [BB, NE]
{
    extern __shared__ char smem[];
    __bf16* wlds  = (__bf16*)smem;                               // [NE][WPITCH] bf16
    float*  slog  = (float*)(smem + W_LDS_BYTES);                // [MT][LOGP]
    float*  s_cnt = (float*)(smem + W_LDS_BYTES + LOG_LDS_BYTES);// [NE]
    float*  s_sum = s_cnt + NE;                                  // [NE]

    const int tid  = threadIdx.x;
    const int wave = tid >> 5;             // 0..3
    const int lane = tid & 31;
    const int l15  = lane & 15;
    const int hi   = lane >> 4;            // 0 for lanes 0-15, 1 for 16-31
    const int tok0 = blockIdx.x * MT;

    // A 16x32 bf16 lane layout (matches B): lanes 0-15 -> K {0..15},
    // lanes 16-31 -> K {16..31}, M = lane&15.
    const int koffA = hi * 16;
    const int koffB = hi * 16;             // B: 16 contiguous K per lane
    const float* xrow = x + (size_t)(tok0 + wave * 16 + l15) * HH;

    v8f acc[4] = {};

    for (int kc0 = 0; kc0 < HH; kc0 += KCH) {
        __syncthreads();   // protect wlds against readers of the previous chunk

        // ---- stage W chunk (f32 global/L2) -> bf16 LDS, RNE via __bf16 cast ----
        for (int i = tid; i < NE * KCH / 4; i += NTHREADS) {
            const int    e = (i * 4) / KCH;
            const int    c = (i * 4) % KCH;
            const float4 v = *(const float4*)(w + (size_t)e * HH + kc0 + c);
            union { __bf16 h[4]; uint2 u; } pk;
            pk.h[0] = (__bf16)v.x; pk.h[1] = (__bf16)v.y;
            pk.h[2] = (__bf16)v.z; pk.h[3] = (__bf16)v.w;
            *(uint2*)(wlds + (size_t)e * WPITCH + c) = pk.u;   // ds_store_b64
        }
        __syncthreads();

        // ---- WMMA over this K chunk: each wave 16 tokens x all 64 experts ----
        for (int kc = 0; kc < KCH; kc += 32) {
            // A fragment: 4 x 16B global loads, convert f32->bf16
            const float* ap = xrow + kc0 + kc;
            const float4 a0 = *(const float4*)(ap + koffA);
            const float4 a1 = *(const float4*)(ap + koffA + 4);
            const float4 a2 = *(const float4*)(ap + koffA + 8);
            const float4 a3 = *(const float4*)(ap + koffA + 12);
            v16bf a;
            a[0]=(__bf16)a0.x;  a[1]=(__bf16)a0.y;  a[2]=(__bf16)a0.z;  a[3]=(__bf16)a0.w;
            a[4]=(__bf16)a1.x;  a[5]=(__bf16)a1.y;  a[6]=(__bf16)a1.z;  a[7]=(__bf16)a1.w;
            a[8]=(__bf16)a2.x;  a[9]=(__bf16)a2.y;  a[10]=(__bf16)a2.z; a[11]=(__bf16)a2.w;
            a[12]=(__bf16)a3.x; a[13]=(__bf16)a3.y; a[14]=(__bf16)a3.z; a[15]=(__bf16)a3.w;

            #pragma unroll
            for (int n = 0; n < 4; ++n) {
                const __bf16* brow = wlds + (size_t)(n * 16 + l15) * WPITCH + kc + koffB;
                v16bf b;
                ((uint4*)&b)[0] = *(const uint4*)brow;        // ds_load_b128
                ((uint4*)&b)[1] = *(const uint4*)(brow + 8);  // ds_load_b128
                acc[n] = __builtin_amdgcn_wmma_f32_16x16x32_bf16(
                    /*neg_a=*/false, a, /*neg_b=*/false, b,
                    /*c_mod=*/(short)0, acc[n],
                    /*reuse_a=*/false, /*reuse_b=*/false);
            }
        }
    }

    // ---- spill accumulators to LDS logits tile ----
    // C/D f32 layout: VGPR v -> M=v (lanes 0-15), M=v+8 (lanes 16-31); N = lane&15.
    #pragma unroll
    for (int n = 0; n < 4; ++n) {
        #pragma unroll
        for (int v = 0; v < 8; ++v) {
            const int m = wave * 16 + v + hi * 8;
            slog[m * LOGP + n * 16 + l15] = acc[n][v];
        }
    }
    if (tid < NE) { s_cnt[tid] = 0.0f; }
    __syncthreads();

    // ---- per-token softmax, overwrite logits row with scores ----
    if (tid < MT) {
        float* row = slog + tid * LOGP;
        float mx = -3.0e38f;
        for (int e = 0; e < NE; ++e) mx = fmaxf(mx, row[e]);
        float s = 0.0f;
        for (int e = 0; e < NE; ++e) s += __expf(row[e] - mx);
        const float inv = 1.0f / s;
        for (int e = 0; e < NE; ++e) row[e] = __expf(row[e] - mx) * inv;
    }
    __syncthreads();

    // ---- per-expert score column sums (no atomics; pad -> conflict-free) ----
    if (tid < NE) {
        float s = 0.0f;
        for (int m = 0; m < MT; ++m) s += slog[m * LOGP + tid];
        s_sum[tid] = s;
    }
    __syncthreads();

    // ---- destructive top-8 + normalize + outputs + count atomics ----
    if (tid < MT) {
        float* row = slog + tid * LOGP;
        const int token = tok0 + tid;
        float wsel[TOPK];
        int   isel[TOPK];
        float tot = 0.0f;
        #pragma unroll
        for (int r = 0; r < TOPK; ++r) {
            float best = -1.0f; int bi = 0;
            for (int e = 0; e < NE; ++e) {
                const float v = row[e];
                if (v > best) { best = v; bi = e; }
            }
            wsel[r] = best; isel[r] = bi; tot += best;
            row[bi] = -1.0f;                      // scores are >= 0
        }
        const float invt = 1.0f / (tot + 1e-20f);
        #pragma unroll
        for (int r = 0; r < TOPK; ++r) {
            out_idx[(size_t)token * TOPK + r] = (float)isel[r];
            out_w[(size_t)token * TOPK + r]   = wsel[r] * invt;
            atomicAdd(&s_cnt[isel[r]], 1.0f);     // ds_add_f32
        }
    }
    __syncthreads();

    // ---- one global atomic per (block, expert) ----
    if (tid < NE) {
        const int b = tok0 / SS;                  // MT | SS, so whole block is one batch
        atomicAdd(&g_counts[b * NE + tid], s_cnt[tid]);
        atomicAdd(&g_sums[b * NE + tid],  s_sum[tid]);
    }
}

// ------------------------------------------------------------------
// finalize: aux = mean_b( sum_e (cnt/(S*K/E)) * (sum/S) ) * ALPHA
// ------------------------------------------------------------------
__global__ __launch_bounds__(256)
void moe_gate_finalize(const float* __restrict__ g_counts,
                       const float* __restrict__ g_sums,
                       float* __restrict__ aux_out)
{
    __shared__ float red[256];
    const int t = threadIdx.x;                    // t indexes (b*NE + e), 256 total
    const float ce = g_counts[t] * (1.0f / ((float)SS * TOPK / NE));   // /512
    const float ms = g_sums[t]   * (1.0f / (float)SS);                 // /4096
    red[t] = ce * ms;
    __syncthreads();
    for (int s = 128; s > 0; s >>= 1) {
        if (t < s) red[t] += red[t + s];
        __syncthreads();
    }
    if (t == 0) aux_out[0] = red[0] * (ALPHA / (float)BB);
}

// ------------------------------------------------------------------
extern "C" void kernel_launch(void* const* d_in, const int* in_sizes, int n_in,
                              void* d_out, int out_size, void* d_ws, size_t ws_size,
                              hipStream_t stream) {
    const float* x = (const float*)d_in[0];   // hidden_states [4,4096,2048] f32
    const float* w = (const float*)d_in[1];   // weight [64,2048] f32

    float* out      = (float*)d_out;
    float* out_idx  = out;                          // [NTOK*TOPK]
    float* out_w    = out + (size_t)NTOK * TOPK;    // [NTOK*TOPK]
    float* aux_out  = out + (size_t)2 * NTOK * TOPK;// [1]

    float* g_counts = (float*)d_ws;                 // [BB*NE]
    float* g_sums   = g_counts + BB * NE;           // [BB*NE]

    // allow >64KB dynamic LDS (84 KB of the 320 KB WGP LDS -> 3 workgroups/WGP)
    static_assert(SMEM_BYTES <= 320 * 1024, "LDS budget");
    (void)hipFuncSetAttribute((const void*)moe_gate_main,
                              hipFuncAttributeMaxDynamicSharedMemorySize,
                              SMEM_BYTES);

    moe_gate_zero<<<1, 256, 0, stream>>>(g_counts);
    moe_gate_main<<<NTOK / MT, NTHREADS, SMEM_BYTES, stream>>>(
        x, w, out_idx, out_w, g_counts, g_sums);
    moe_gate_finalize<<<1, 256, 0, stream>>>(g_counts, g_sums, aux_out);
}
